// SamCNN_66357244723770
// MI455X (gfx1250) — matrix operand order repaired and checked by reading, not compiled, not run on hardware
//
#include <hip/hip_runtime.h>
#include <hip/hip_bf16.h>
#include <math.h>

// ---------------- CDNA5 WMMA types ----------------
typedef __bf16 bf16_t;
typedef __bf16 v16bf __attribute__((ext_vector_type(16)));
typedef __bf16 v8bf  __attribute__((ext_vector_type(8)));
typedef float  v8f   __attribute__((ext_vector_type(8)));
typedef int    v4i   __attribute__((ext_vector_type(4)));

// Problem constants (match reference)
#define BB   32
#define QQ   48
#define AA   48
#define DD   300
#define DP   320      // D padded to multiple of 32 (bf16 WMMA k-step)
#define CC   512
#define KK   5
#define HH   512
#define PP   512
#define LOUT 52       // Q + K - 1
#define NKC  2560     // K*C columns of the "G" GEMM (ordered k*512 + c)

// ---------------- gfx1250 async global->LDS path ----------------
#if __has_builtin(__builtin_amdgcn_global_load_async_to_lds_b128)
#define HAVE_ASYNC_LDS 1
#endif

__device__ __forceinline__ void cp16_g2l(const bf16_t* g, bf16_t* l) {
#ifdef HAVE_ASYNC_LDS
  // signature: (int4 addrspace(1)*, int4 addrspace(3)*, imm offset, imm cpol)
  __builtin_amdgcn_global_load_async_to_lds_b128(
      (__attribute__((address_space(1))) v4i*)(v4i*)(void*)g,
      (__attribute__((address_space(3))) v4i*)(v4i*)(void*)l,
      0, 0);
#else
  *(uint4*)l = *(const uint4*)g;
#endif
}

__device__ __forceinline__ void cp_join() {
#ifdef HAVE_ASYNC_LDS
#if __has_builtin(__builtin_amdgcn_s_wait_asynccnt)
  __builtin_amdgcn_s_wait_asynccnt(0);   // drain this wave's ASYNCcnt before barrier
#endif
#endif
  __syncthreads();
}

// =====================================================================
// Conversion kernels: fp32 -> bf16 (with K padding / filter repack)
// =====================================================================
__global__ void cvt_pad_kernel(const float* __restrict__ src, bf16_t* __restrict__ dst,
                               int rows, int sk, int dk) {
  int n = rows * dk;
  for (int idx = blockIdx.x * blockDim.x + threadIdx.x; idx < n; idx += gridDim.x * blockDim.x) {
    int r = idx / dk, d = idx - r * dk;
    dst[idx] = (bf16_t)((d < sk) ? src[(size_t)r * sk + d] : 0.0f);
  }
}

// Conv weights (C,1,K,D) -> bf16 [2560, 320] rows ordered j = k*512 + c, zero-padded d>=300
__global__ void cvt_w_kernel(const float* __restrict__ src, bf16_t* __restrict__ dst) {
  int n = NKC * DP;
  for (int idx = blockIdx.x * blockDim.x + threadIdx.x; idx < n; idx += gridDim.x * blockDim.x) {
    int j = idx / DP, d = idx - j * DP;
    int k = j >> 9, c = j & 511;
    dst[idx] = (bf16_t)((d < DD) ? src[(size_t)(c * KK + k) * DD + d] : 0.0f);
  }
}

__global__ void cvt_kernel(const float* __restrict__ src, bf16_t* __restrict__ dst, int n) {
  for (int i = blockIdx.x * blockDim.x + threadIdx.x; i < n; i += gridDim.x * blockDim.x)
    dst[i] = (bf16_t)src[i];
}

// =====================================================================
// Cosine similarity: cos[b,i,j] = q_i . a_j / max(|q_i||a_j|, 1e-6)
// =====================================================================
__global__ void cos_kernel(const float* __restrict__ q, const float* __restrict__ a,
                           float* __restrict__ cosm) {
  int idx = blockIdx.x * blockDim.x + threadIdx.x;
  if (idx >= BB * QQ * AA) return;
  int j = idx % AA, i = (idx / AA) % QQ, b = idx / (QQ * AA);
  const float* qr = q + (size_t)(b * QQ + i) * DD;
  const float* ar = a + (size_t)(b * AA + j) * DD;
  float qq = 0.f, aa = 0.f, qa = 0.f;
  for (int d = 0; d < DD; ++d) {
    float x = qr[d], y = ar[d];
    qq = fmaf(x, x, qq); aa = fmaf(y, y, aa); qa = fmaf(x, y, qa);
  }
  cosm[idx] = qa / fmaxf(sqrtf(qq) * sqrtf(aa), 1e-6f);
}

// =====================================================================
// Fragment helpers (CDNA5 ISA 7.12.2 layouts)
//   A 16x32 bf16: lanes 0-15 -> K {8hi+0..7, 16+8hi+0..7}
//   B 32x16 bf16 (from B^T rows): same K order as A
// =====================================================================
__device__ __forceinline__ v16bf frag_from_lds(const bf16_t* row, int hi) {
  v8bf lo = *(const v8bf*)(row + 8 * hi);
  v8bf up = *(const v8bf*)(row + 16 + 8 * hi);
  return __builtin_shufflevector(lo, up, 0, 1, 2, 3, 4, 5, 6, 7,
                                 8, 9, 10, 11, 12, 13, 14, 15);
}
__device__ __forceinline__ v16bf fragB_from_lds(const bf16_t* row, int hi) {
  v8bf lo = *(const v8bf*)(row + 8 * hi);
  v8bf up = *(const v8bf*)(row + 16 + 8 * hi);
  return __builtin_shufflevector(lo, up, 0, 1, 2, 3, 4, 5, 6, 7,
                                 8, 9, 10, 11, 12, 13, 14, 15);
}

// =====================================================================
// Big WMMA GEMM: C[M,N] = A[M,K] * B[N,K]^T (+bias). M % 64 == 0, N % 128 == 0,
// K % 32 == 0. Block = 256 thr = 8 waves (2Mx4N), wave computes 32x32 (4 accum).
// Tiles staged via GLOBAL_LOAD_ASYNC_TO_LDS (ASYNCcnt) when available.
// =====================================================================
#define BM 64
#define BN 128
#define TPAD 40   // LDS row pitch in elements (80 B) to spread banks
__global__ __launch_bounds__(256) void gemm_bt_wmma_big(const bf16_t* __restrict__ A,
                                                        const bf16_t* __restrict__ B,
                                                        const float* __restrict__ bias,
                                                        float* __restrict__ C,
                                                        int M, int N, int K) {
  __shared__ __align__(16) bf16_t tA[BM * TPAD];
  __shared__ __align__(16) bf16_t tB[BN * TPAD];
  int tid = threadIdx.x;
  int lane = tid & 31, wave = tid >> 5;
  int wm = wave >> 2, wn = wave & 3;
  int mb = blockIdx.y * BM, nb = blockIdx.x * BN;
  int hi = lane >> 4, rc = lane & 15;

  // copy slot assignment: 16 B per slot
  int ar = tid >> 2, ac = (tid & 3) * 8;           // A: 64 rows x 4 chunks
  const bf16_t* Ag = A + (size_t)(mb + ar) * K + ac;
  bf16_t* Al = &tA[ar * TPAD + ac];

  v8f acc00 = {}, acc01 = {}, acc10 = {}, acc11 = {};
  for (int kk = 0; kk < K; kk += 32) {
    cp16_g2l(Ag + kk, Al);
#pragma unroll
    for (int s = 0; s < 2; ++s) {                  // B: 128 rows x 4 chunks = 512 slots
      int cid = tid * 2 + s;
      int r = cid >> 2, kc = (cid & 3) * 8;
      cp16_g2l(B + (size_t)(nb + r) * K + kk + kc, &tB[r * TPAD + kc]);
    }
    cp_join();

    v16bf af0 = frag_from_lds(&tA[(wm * 32 + rc) * TPAD], hi);
    v16bf af1 = frag_from_lds(&tA[(wm * 32 + 16 + rc) * TPAD], hi);
    v16bf bf0 = fragB_from_lds(&tB[(wn * 32 + rc) * TPAD], hi);
    v16bf bf1 = fragB_from_lds(&tB[(wn * 32 + 16 + rc) * TPAD], hi);

    acc00 = __builtin_amdgcn_wmma_f32_16x16x32_bf16(false, af0, false, bf0, (short)0, acc00, false, false);
    acc01 = __builtin_amdgcn_wmma_f32_16x16x32_bf16(false, af0, false, bf1, (short)0, acc01, false, false);
    acc10 = __builtin_amdgcn_wmma_f32_16x16x32_bf16(false, af1, false, bf0, (short)0, acc10, false, false);
    acc11 = __builtin_amdgcn_wmma_f32_16x16x32_bf16(false, af1, false, bf1, (short)0, acc11, false, false);
    __syncthreads();   // protect tiles before next stage overwrites
  }

  int colbase = nb + wn * 32;
  int rowbase = mb + wm * 32;
#pragma unroll
  for (int tj = 0; tj < 2; ++tj) {
    int col = colbase + tj * 16 + rc;
    float bv = bias ? bias[col] : 0.0f;
    v8f acc_t0 = tj ? acc01 : acc00;
    v8f acc_t1 = tj ? acc11 : acc10;
#pragma unroll
    for (int i = 0; i < 8; ++i) {
      int r0 = rowbase + i + 8 * hi;
      C[(size_t)r0 * N + col] = acc_t0[i] + bv;
      C[(size_t)(r0 + 16) * N + col] = acc_t1[i] + bv;
    }
  }
}

// =====================================================================
// Small WMMA GEMM (direct loads): used for M == 32 layers.
// =====================================================================
__global__ __launch_bounds__(256) void gemm_bt_wmma(const bf16_t* __restrict__ A,
                                                    const bf16_t* __restrict__ B,
                                                    const float* __restrict__ bias,
                                                    float* __restrict__ C,
                                                    int M, int N, int K) {
  int lane = threadIdx.x & 31;
  int wave = threadIdx.x >> 5;
  int n0 = (blockIdx.x * 8 + wave) * 16;
  int m0 = blockIdx.y * 16;
  if (n0 >= N || m0 >= M) return;            // wave-uniform
  int hi = lane >> 4;
  int rc = lane & 15;
  const bf16_t* Ap = A + (size_t)(m0 + rc) * K;
  const bf16_t* Bp = B + (size_t)(n0 + rc) * K;

  v8f acc = {};
  for (int kk = 0; kk < K; kk += 32) {
    v16bf af, bf;
#pragma unroll
    for (int i = 0; i < 8; ++i) {
      int kb = ((i < 4) ? 2 * i : 16 + 2 * (i - 4)) + 8 * hi;
      af[2 * i]     = Ap[kk + kb];
      af[2 * i + 1] = Ap[kk + kb + 1];
    }
#pragma unroll
    for (int i = 0; i < 8; ++i) {
      int kb = ((i < 4) ? 2 * i : 16 + 2 * (i - 4)) + 8 * hi;
      bf[2 * i]     = Bp[kk + kb];
      bf[2 * i + 1] = Bp[kk + kb + 1];
    }
    acc = __builtin_amdgcn_wmma_f32_16x16x32_bf16(false, af, false, bf,
                                                  (short)0, acc, false, false);
  }
  int col = n0 + rc;
  float bv = bias ? bias[col] : 0.0f;
#pragma unroll
  for (int i = 0; i < 8; ++i) {
    int row = m0 + i + 8 * hi;
    C[(size_t)row * N + col] = acc[i] + bv;
  }
}

// =====================================================================
// sm-path combine: pooled[b,c] = max(0, max_t (bias_c + sum_k G[b, t-4+k, k, c]))
// =====================================================================
__global__ void sm_combine_kernel(const float* __restrict__ G, const float* __restrict__ bias,
                                  bf16_t* __restrict__ poolbf) {
  int idx = blockIdx.x * blockDim.x + threadIdx.x;
  if (idx >= BB * CC) return;
  int b = idx >> 9, c = idx & 511;
  float bc = bias[c];
  float m = -1e30f;
  for (int t = 0; t < LOUT; ++t) {
    float s = bc;
#pragma unroll
    for (int k = 0; k < KK; ++k) {
      int j = t - 4 + k;
      if ((unsigned)j < (unsigned)QQ)
        s += G[(size_t)(b * QQ + j) * NKC + (k << 9) + c];
    }
    m = fmaxf(m, s);
  }
  poolbf[idx] = (bf16_t)fmaxf(m, 0.0f);
}

// =====================================================================
// attn combine: pooled[bq,c] = max(0, max_t (bias_c + sum_k cos[b,q,t-4+k]*G[b,t-4+k,k,c]))
// grid = (C/64, Q/8, B); G tile + cos rows staged in LDS.
// =====================================================================
#define CT 64
__global__ void attn_combine_kernel(const float* __restrict__ G, const float* __restrict__ cosm,
                                    const float* __restrict__ bias, bf16_t* __restrict__ poolbf) {
  extern __shared__ float smem[];
  float* Gs   = smem;                 // [48*5*CT]
  float* cosS = smem + AA * KK * CT;  // [8*48]
  int c0 = blockIdx.x * CT;
  int qg = blockIdx.y;
  int b  = blockIdx.z;

  for (int i = threadIdx.x; i < AA * KK * CT; i += blockDim.x) {
    int cc = i % CT, jk = i / CT;
    int k = jk % KK, j = jk / KK;
    Gs[i] = G[(size_t)(b * AA + j) * NKC + (k << 9) + c0 + cc];
  }
  for (int i = threadIdx.x; i < 8 * AA; i += blockDim.x) {
    int qi = qg * 8 + i / AA;
    cosS[i] = cosm[(size_t)(b * QQ + qi) * AA + (i % AA)];
  }
  __syncthreads();

  for (int w = threadIdx.x; w < 8 * CT; w += blockDim.x) {
    int cc = w % CT, ql = w / CT;
    const float* cr = cosS + ql * AA;
    float bc = bias[c0 + cc];
    float m = -1e30f;
    for (int t = 0; t < LOUT; ++t) {
      float s = bc;
#pragma unroll
      for (int k = 0; k < KK; ++k) {
        int j = t - 4 + k;
        if ((unsigned)j < (unsigned)AA)
          s = fmaf(cr[j], Gs[(j * KK + k) * CT + cc], s);
      }
      m = fmaxf(m, s);
    }
    int qi = qg * 8 + ql;
    poolbf[(size_t)(b * QQ + qi) * CC + c0 + cc] = (bf16_t)fmaxf(m, 0.0f);
  }
}

// =====================================================================
__global__ void mean_kernel(const float* __restrict__ fcout, float* __restrict__ feat) {
  int idx = blockIdx.x * blockDim.x + threadIdx.x;
  if (idx >= BB * HH) return;
  int b = idx >> 9, h = idx & 511;
  float s = 0.f;
  for (int qi = 0; qi < QQ; ++qi) s += fcout[(size_t)(b * QQ + qi) * HH + h];
  feat[idx] = s * (1.0f / QQ);
}

__global__ void concat_kernel(const float* __restrict__ ac, const float* __restrict__ bc,
                              const float* __restrict__ af, bf16_t* __restrict__ comb) {
  int idx = blockIdx.x * blockDim.x + threadIdx.x;
  if (idx >= BB * 3 * HH) return;
  int b = idx / (3 * HH), i = idx - b * 3 * HH;
  float v = (i < HH) ? ac[b * HH + i]
          : (i < 2 * HH) ? bc[b * HH + i - HH]
          : af[b * HH + i - 2 * HH];
  comb[idx] = (bf16_t)v;
}

__global__ void bn_tanh_kernel(const float* __restrict__ x, const float* __restrict__ gamma,
                               const float* __restrict__ beta, float* __restrict__ y) {
  int p = blockIdx.x * blockDim.x + threadIdx.x;
  if (p >= PP) return;
  float mu = 0.f;
  for (int b = 0; b < BB; ++b) mu += x[(size_t)b * PP + p];
  mu *= (1.0f / BB);
  float var = 0.f;
  for (int b = 0; b < BB; ++b) { float d = x[(size_t)b * PP + p] - mu; var = fmaf(d, d, var); }
  var *= (1.0f / BB);
  float inv = 1.0f / sqrtf(var + 1e-5f);
  float g = gamma[p], be = beta[p];
  for (int b = 0; b < BB; ++b)
    y[(size_t)b * PP + p] = tanhf((x[(size_t)b * PP + p] - mu) * inv * g + be);
}

__global__ void out_kernel(const float* __restrict__ t, const float* __restrict__ w,
                           const float* __restrict__ bo, float* __restrict__ out) {
  int idx = blockIdx.x * blockDim.x + threadIdx.x;
  if (idx >= BB * 2) return;
  int b = idx >> 1, cl = idx & 1;
  float s = bo[cl];
  for (int p = 0; p < PP; ++p) s = fmaf(t[(size_t)b * PP + p], w[(size_t)cl * PP + p], s);
  out[idx] = s;
}

// =====================================================================
extern "C" void kernel_launch(void* const* d_in, const int* in_sizes, int n_in,
                              void* d_out, int out_size, void* d_ws, size_t ws_size,
                              hipStream_t stream) {
  const float* qf        = (const float*)d_in[0];   // a_features == q
  const float* af        = (const float*)d_in[1];   // b_features == a
  const float* sm_conv_w = (const float*)d_in[2];
  const float* sm_conv_b = (const float*)d_in[3];
  const float* sm_fc_w   = (const float*)d_in[4];
  const float* sm_fc_b   = (const float*)d_in[5];
  const float* ctx_conv_w= (const float*)d_in[6];
  const float* ctx_conv_b= (const float*)d_in[7];
  const float* ctx_fc_w  = (const float*)d_in[8];
  const float* ctx_fc_b  = (const float*)d_in[9];
  const float* proj_w    = (const float*)d_in[10];
  const float* proj_b    = (const float*)d_in[11];
  const float* bn_gamma  = (const float*)d_in[12];
  const float* bn_beta   = (const float*)d_in[13];
  const float* out_w     = (const float*)d_in[14];
  const float* out_b     = (const float*)d_in[15];
  float* out = (float*)d_out;

  char* ws = (char*)d_ws;
  size_t off = 0;
  auto alloc = [&](size_t bytes) { off = (off + 255) & ~(size_t)255; size_t o = off; off += bytes; return o; };

  const int RQ = BB * QQ;  // 1536 rows
  bf16_t* q_bf     = (bf16_t*)(ws + alloc((size_t)RQ * DP * 2));
  bf16_t* a_bf     = (bf16_t*)(ws + alloc((size_t)RQ * DP * 2));
  bf16_t* wsm_bf   = (bf16_t*)(ws + alloc((size_t)NKC * DP * 2));
  bf16_t* wctx_bf  = (bf16_t*)(ws + alloc((size_t)NKC * DP * 2));
  bf16_t* smfc_bf  = (bf16_t*)(ws + alloc((size_t)HH * CC * 2));
  bf16_t* ctxfc_bf = (bf16_t*)(ws + alloc((size_t)HH * CC * 2));
  bf16_t* projw_bf = (bf16_t*)(ws + alloc((size_t)PP * 3 * HH * 2));
  float*  cosm     = (float*) (ws + alloc((size_t)BB * QQ * AA * 4));
  float*  Gq       = (float*) (ws + alloc((size_t)RQ * NKC * 4));
  float*  Ga       = (float*) (ws + alloc((size_t)RQ * NKC * 4));
  float*  Gctx     = (float*) (ws + alloc((size_t)RQ * NKC * 4));
  bf16_t* poolq_bf = (bf16_t*)(ws + alloc((size_t)BB * CC * 2));
  bf16_t* poola_bf = (bf16_t*)(ws + alloc((size_t)BB * CC * 2));
  bf16_t* poolat_bf= (bf16_t*)(ws + alloc((size_t)RQ * CC * 2));
  float*  a_cnn    = (float*) (ws + alloc((size_t)BB * HH * 4));
  float*  b_cnn    = (float*) (ws + alloc((size_t)BB * HH * 4));
  float*  attnfc   = (float*) (ws + alloc((size_t)RQ * HH * 4));
  float*  attnfeat = (float*) (ws + alloc((size_t)BB * HH * 4));
  bf16_t* comb_bf  = (bf16_t*)(ws + alloc((size_t)BB * 3 * HH * 2));
  float*  projout  = (float*) (ws + alloc((size_t)BB * PP * 4));
  float*  tanhv    = (float*) (ws + alloc((size_t)BB * PP * 4));
  (void)ws_size; (void)in_sizes; (void)n_in; (void)out_size;

  const int TB = 256;
  // --- bf16 conversions / repacks ---
  cvt_pad_kernel<<<(RQ * DP + TB - 1) / TB, TB, 0, stream>>>(qf, q_bf, RQ, DD, DP);
  cvt_pad_kernel<<<(RQ * DP + TB - 1) / TB, TB, 0, stream>>>(af, a_bf, RQ, DD, DP);
  cvt_w_kernel<<<(NKC * DP + TB - 1) / TB, TB, 0, stream>>>(sm_conv_w, wsm_bf);
  cvt_w_kernel<<<(NKC * DP + TB - 1) / TB, TB, 0, stream>>>(ctx_conv_w, wctx_bf);
  cvt_kernel<<<(HH * CC + TB - 1) / TB, TB, 0, stream>>>(sm_fc_w, smfc_bf, HH * CC);
  cvt_kernel<<<(HH * CC + TB - 1) / TB, TB, 0, stream>>>(ctx_fc_w, ctxfc_bf, HH * CC);
  cvt_kernel<<<(PP * 3 * HH + TB - 1) / TB, TB, 0, stream>>>(proj_w, projw_bf, PP * 3 * HH);

  // --- cosine attention matrix ---
  cos_kernel<<<(BB * QQ * AA + TB - 1) / TB, TB, 0, stream>>>(qf, af, cosm);

  // --- the three "G" GEMMs: [1536,320] x [2560,320]^T (async-LDS + WMMA) ---
  gemm_bt_wmma_big<<<dim3(NKC / BN, RQ / BM), TB, 0, stream>>>(q_bf, wsm_bf,  nullptr, Gq,   RQ, NKC, DP);
  gemm_bt_wmma_big<<<dim3(NKC / BN, RQ / BM), TB, 0, stream>>>(a_bf, wsm_bf,  nullptr, Ga,   RQ, NKC, DP);
  gemm_bt_wmma_big<<<dim3(NKC / BN, RQ / BM), TB, 0, stream>>>(a_bf, wctx_bf, nullptr, Gctx, RQ, NKC, DP);

  // --- conv reconstruction + relu + maxpool ---
  sm_combine_kernel<<<(BB * CC + TB - 1) / TB, TB, 0, stream>>>(Gq, sm_conv_b, poolq_bf);
  sm_combine_kernel<<<(BB * CC + TB - 1) / TB, TB, 0, stream>>>(Ga, sm_conv_b, poola_bf);
  size_t shmem = (size_t)(AA * KK * CT + 8 * AA) * 4;   // 62,976 B < 64 KB
  attn_combine_kernel<<<dim3(CC / CT, QQ / 8, BB), TB, shmem, stream>>>(Gctx, cosm, ctx_conv_b, poolat_bf);

  // --- FC layers ---
  gemm_bt_wmma<<<dim3(HH / 128, BB / 16), TB, 0, stream>>>(poolq_bf, smfc_bf, sm_fc_b, a_cnn, BB, HH, CC);
  gemm_bt_wmma<<<dim3(HH / 128, BB / 16), TB, 0, stream>>>(poola_bf, smfc_bf, sm_fc_b, b_cnn, BB, HH, CC);
  gemm_bt_wmma_big<<<dim3(HH / BN, RQ / BM), TB, 0, stream>>>(poolat_bf, ctxfc_bf, ctx_fc_b, attnfc, RQ, HH, CC);

  mean_kernel<<<(BB * HH + TB - 1) / TB, TB, 0, stream>>>(attnfc, attnfeat);
  concat_kernel<<<(BB * 3 * HH + TB - 1) / TB, TB, 0, stream>>>(a_cnn, b_cnn, attnfeat, comb_bf);

  // --- projection, batchnorm + tanh, output head ---
  gemm_bt_wmma<<<dim3(PP / 128, BB / 16), TB, 0, stream>>>(comb_bf, projw_bf, proj_b, projout, BB, PP, 3 * HH);
  bn_tanh_kernel<<<(PP + TB - 1) / TB, TB, 0, stream>>>(projout, bn_gamma, bn_beta, tanhv);
  out_kernel<<<1, 64, 0, stream>>>(tanhv, out_w, out_b, out);
}
